// biLSTM_13280038879416
// MI455X (gfx1250) — matrix operand order repaired and checked by reading, not compiled, not run on hardware
//
#include <hip/hip_runtime.h>
#include <hip/hip_bf16.h>

// ---------------------------------------------------------------------------
// BiLSTM (2-layer, H=64, B=512, T=128) + FC for gfx1250 (MI455X).
// bf16 WMMA (v_wmma_f32_16x16x32_bf16) for all GEMMs, fp32 accumulation.
// Round 2: dual-direction fused scan (halves sequential critical path),
// software-pipelined gate prefetch in the scan, 4 n-tiles/wave in the GEMM.
// ---------------------------------------------------------------------------

typedef __attribute__((ext_vector_type(16))) __bf16 v16bf;
typedef __attribute__((ext_vector_type(8)))  float  v8f;

static constexpr int HID    = 64;
static constexpr int SEQ_T  = 128;
static constexpr int BATCH  = 512;
static constexpr int MTOT   = BATCH * SEQ_T;   // 65536 rows
static constexpr int NGATES = 512;             // 4*H * 2 directions, concatenated
static constexpr int K0PAD  = 320;             // 300 padded to multiple of 32
static constexpr int K1     = 128;             // 2*H

union FragBf { uint4 u4[2]; v16bf v; };        // 16 bf16 halves = 8 VGPRs
union AccF   { float f[8];  v8f  v; };         // 16x16 f32 C/D fragment

__device__ __forceinline__ unsigned short f2bf(float x) {
  union { float f; unsigned u; } v; v.f = x;
  unsigned r = v.u + 0x7FFFu + ((v.u >> 16) & 1u);   // round-to-nearest-even
  return (unsigned short)(r >> 16);
}
__device__ __forceinline__ float sigm(float x) { return 1.0f / (1.0f + __expf(-x)); }

// --- pack fp32 matrix -> bf16 (row-major), zero-padding columns -------------
__global__ void pack_bf16(const float* __restrict__ src, unsigned short* __restrict__ dst,
                          int src_cols, int dst_cols) {
  int r = blockIdx.x, c = threadIdx.x;
  float v = (c < src_cols) ? src[(size_t)r * src_cols + c] : 0.0f;
  dst[(size_t)r * dst_cols + c] = f2bf(v);
}

// --- concat two 256-float bias vectors --------------------------------------
__global__ void concat_bias(const float* __restrict__ a, const float* __restrict__ b,
                            float* __restrict__ dst) {
  int i = blockIdx.x * blockDim.x + threadIdx.x;
  if (i < 256) dst[i] = a[i];
  else if (i < 512) dst[i] = b[i - 256];
}

// --- embedding gather -> bf16, K padded 300 -> 320 --------------------------
__global__ __launch_bounds__(320)
void embed_gather(const int* __restrict__ x, const float* __restrict__ emb,
                  unsigned short* __restrict__ X0) {
  int m = blockIdx.x;          // b*T + t
  int k = threadIdx.x;
  int tok = x[m];
  float v = (k < 300) ? emb[(size_t)tok * 300 + k] : 0.0f;
  X0[(size_t)m * K0PAD + k] = f2bf(v);
}

// --- big gate GEMM: OUT[M x N] = X[M x Kpad](bf16) * W[N x Kpad]^T + bias ---
// block = 128 threads = 4 waves; each wave computes a 16x64 output tile
// (4 independent WMMA accumulator chains share each A fragment).
__global__ __launch_bounds__(128)
void wmma_gemm_bias(const unsigned short* __restrict__ X, const unsigned short* __restrict__ W,
                    const float* __restrict__ bias, float* __restrict__ OUT,
                    int Kpad, int N) {
  const int lane = threadIdx.x & 31;
  const int wave = threadIdx.x >> 5;
  const int m0   = blockIdx.y * 16;
  const int n0   = blockIdx.x * 256 + wave * 64;
  const int nidx = lane & 15;        // C-layout column, A-layout row
  const int lhi  = lane >> 4;        // lane-half selector

  AccF acc[4];
#pragma unroll
  for (int nt = 0; nt < 4; ++nt) {
    float bv = bias[n0 + nt * 16 + nidx];
#pragma unroll
    for (int i = 0; i < 8; ++i) acc[nt].f[i] = bv;
  }

  const int ktiles = Kpad >> 5;
  for (int kt = 0; kt < ktiles; ++kt) {
    const int kb = kt * 32;
    FragBf a;   // A 16x32: lane row = nidx, K offset = lhi*8 (and +16)
    const unsigned short* xr = X + (size_t)(m0 + nidx) * Kpad + kb + lhi * 8;
    a.u4[0] = *(const uint4*)(xr);
    a.u4[1] = *(const uint4*)(xr + 16);
#pragma unroll
    for (int nt = 0; nt < 4; ++nt) {
      FragBf b; // B 32x16: lane col = nidx, K offset = lhi*8 (and +16)
      const unsigned short* wr = W + (size_t)(n0 + nt * 16 + nidx) * Kpad + kb + lhi * 8;
      b.u4[0] = *(const uint4*)(wr);
      b.u4[1] = *(const uint4*)(wr + 16);
      acc[nt].v = __builtin_amdgcn_wmma_f32_16x16x32_bf16(
          false, a.v, false, b.v, (short)0, acc[nt].v, false, false);
    }
  }
#pragma unroll
  for (int nt = 0; nt < 4; ++nt)
#pragma unroll
    for (int i = 0; i < 8; ++i)
      OUT[(size_t)(m0 + i + 8 * lhi) * N + n0 + nt * 16 + nidx] = acc[nt].f[i];
}

// --- LSTM recurrence, both directions fused (blockIdx.y = direction) --------
// One workgroup = 16 batch rows x 1 direction, 4 waves; wave w owns hidden
// slice j0=16w and computes gates i,f,g,o for it (8 WMMAs/step). Cell state
// lives in VGPRs; h (16x64 bf16) cycles through LDS; next step's gate biases
// are prefetched into registers while the current step computes.
__global__ __launch_bounds__(128)
void lstm_scan_dual(const float* __restrict__ GX,            // stride NGATES; dir d at +d*256
                    const unsigned short* __restrict__ WhhF, // 256 x 64 bf16 (forward)
                    const unsigned short* __restrict__ WhhR, // 256 x 64 bf16 (reverse)
                    unsigned short* __restrict__ Hall,       // or null; dir d at +d*64, stride 128
                    float* __restrict__ Hfinal,              // or null; dir d at +d*64, stride 128
                    int nsteps_f, int nsteps_r) {
  __shared__ unsigned short Hlds[16 * HID];
  const int dir    = blockIdx.y;                 // 0 = forward, 1 = reverse
  const int nsteps = dir ? nsteps_r : nsteps_f;
  const unsigned short* Whh = dir ? WhhR : WhhF;
  const float* gx = GX + dir * 256;
  unsigned short* hall = Hall ? (Hall + dir * 64) : nullptr;
  float* hfin = Hfinal ? (Hfinal + dir * 64) : nullptr;

  const int lane = threadIdx.x & 31;
  const int wave = threadIdx.x >> 5;
  const int j0   = wave * 16;
  const int b0   = blockIdx.x * 16;
  const int nidx = lane & 15;
  const int lhi  = lane >> 4;

  // Preload W_hh fragments for this wave's 4 gate tiles x 2 k-tiles.
  FragBf Bf[4][2];
#pragma unroll
  for (int gi = 0; gi < 4; ++gi)
#pragma unroll
    for (int kt = 0; kt < 2; ++kt) {
      const unsigned short* wr = Whh + (size_t)(gi * HID + j0 + nidx) * HID + kt * 32 + lhi * 8;
      Bf[gi][kt].u4[0] = *(const uint4*)(wr);
      Bf[gi][kt].u4[1] = *(const uint4*)(wr + 16);
    }

  for (int i = threadIdx.x; i < 16 * HID; i += blockDim.x) Hlds[i] = 0;
  __syncthreads();

  AccF cst;                       // cell state, persistent in VGPRs
#pragma unroll
  for (int i = 0; i < 8; ++i) cst.f[i] = 0.0f;
  float hlast[8];
#pragma unroll
  for (int i = 0; i < 8; ++i) hlast[i] = 0.0f;

  // Software pipeline: gpre holds gate biases (Wx+b part) for the *next* step.
  AccF gpre[4];
  {
    const int t0 = dir ? (SEQ_T - 1) : 0;
#pragma unroll
    for (int gi = 0; gi < 4; ++gi)
#pragma unroll
      for (int i = 0; i < 8; ++i)
        gpre[gi].f[i] =
            gx[((size_t)(b0 + i + 8 * lhi) * SEQ_T + t0) * NGATES + gi * HID + j0 + nidx];
  }

  for (int s = 0; s < nsteps; ++s) {
    const int t = dir ? (SEQ_T - 1 - s) : s;

    FragBf A[2];                  // h tile 16x64 -> two 16x32 A fragments
#pragma unroll
    for (int kt = 0; kt < 2; ++kt) {
      const unsigned short* hr = Hlds + nidx * HID + kt * 32 + lhi * 8;
      A[kt].u4[0] = *(const uint4*)(hr);
      A[kt].u4[1] = *(const uint4*)(hr + 16);
    }
    __syncthreads();              // all waves read H before it is overwritten

    // Consume prefetched gate biases; immediately issue next step's loads.
    AccF g[4];
#pragma unroll
    for (int gi = 0; gi < 4; ++gi) g[gi].v = gpre[gi].v;

    const int sn = (s + 1 < nsteps) ? (s + 1) : s;
    const int tn = dir ? (SEQ_T - 1 - sn) : sn;
#pragma unroll
    for (int gi = 0; gi < 4; ++gi)
#pragma unroll
      for (int i = 0; i < 8; ++i)
        gpre[gi].f[i] =
            gx[((size_t)(b0 + i + 8 * lhi) * SEQ_T + tn) * NGATES + gi * HID + j0 + nidx];

#pragma unroll
    for (int gi = 0; gi < 4; ++gi)
#pragma unroll
      for (int kt = 0; kt < 2; ++kt)
        g[gi].v = __builtin_amdgcn_wmma_f32_16x16x32_bf16(
            false, A[kt].v, false, Bf[gi][kt].v, (short)0, g[gi].v, false, false);

#pragma unroll
    for (int i = 0; i < 8; ++i) {
      float iv = sigm(g[0].f[i]);
      float fv = sigm(g[1].f[i]);
      float gv = tanhf(g[2].f[i]);
      float ov = sigm(g[3].f[i]);
      float c  = fv * cst.f[i] + iv * gv;
      float h  = ov * tanhf(c);
      cst.f[i]  = c;
      hlast[i]  = h;
      int mr = i + 8 * lhi;
      Hlds[mr * HID + j0 + nidx] = f2bf(h);
      if (hall)
        hall[((size_t)(b0 + mr) * SEQ_T + t) * 128 + j0 + nidx] = f2bf(h);
    }
    __syncthreads();              // new H visible before next step's A load
  }

  if (hfin)
#pragma unroll
    for (int i = 0; i < 8; ++i)
      hfin[(size_t)(b0 + i + 8 * lhi) * 128 + j0 + nidx] = hlast[i];
}

// --- tiny FC: out[512x12] = last[512x128] * fc_w[12x128]^T + fc_b -----------
__global__ __launch_bounds__(128)
void fc_kernel(const float* __restrict__ last, const float* __restrict__ w,
               const float* __restrict__ bias, float* __restrict__ out) {
  int idx = blockIdx.x * blockDim.x + threadIdx.x;
  if (idx >= BATCH * 12) return;
  int b = idx / 12, o = idx % 12;
  float s = bias[o];
#pragma unroll 8
  for (int k = 0; k < 128; ++k) s += last[b * 128 + k] * w[o * 128 + k];
  out[idx] = s;
}

// ---------------------------------------------------------------------------
extern "C" void kernel_launch(void* const* d_in, const int* in_sizes, int n_in,
                              void* d_out, int out_size, void* d_ws, size_t ws_size,
                              hipStream_t stream) {
  const int*   x        = (const int*)  d_in[0];
  const float* emb      = (const float*)d_in[1];
  const float* w_ih_l0f = (const float*)d_in[2];
  const float* w_hh_l0f = (const float*)d_in[3];
  const float* b_l0f    = (const float*)d_in[4];
  const float* w_ih_l0r = (const float*)d_in[5];
  const float* w_hh_l0r = (const float*)d_in[6];
  const float* b_l0r    = (const float*)d_in[7];
  const float* w_ih_l1f = (const float*)d_in[8];
  const float* w_hh_l1f = (const float*)d_in[9];
  const float* b_l1f    = (const float*)d_in[10];
  const float* w_ih_l1r = (const float*)d_in[11];
  const float* w_hh_l1r = (const float*)d_in[12];
  const float* b_l1r    = (const float*)d_in[13];
  const float* fc_w     = (const float*)d_in[14];
  const float* fc_b     = (const float*)d_in[15];
  float* out = (float*)d_out;

  // Workspace layout (256B-aligned regions)
  char* ws = (char*)d_ws;
  size_t off = 0;
  auto carve = [&](size_t bytes) { char* p = ws + off; off = (off + bytes + 255) & ~(size_t)255; return p; };
  unsigned short* X0    = (unsigned short*)carve((size_t)MTOT * K0PAD * 2);  // 42 MB
  float*          GXb   = (float*)         carve((size_t)MTOT * NGATES * 4); // 134 MB, reused L0/L1
  unsigned short* X1    = (unsigned short*)carve((size_t)MTOT * K1 * 2);     // 17 MB
  unsigned short* W0    = (unsigned short*)carve((size_t)NGATES * K0PAD * 2);
  unsigned short* W1    = (unsigned short*)carve((size_t)NGATES * K1 * 2);
  unsigned short* Whh0f = (unsigned short*)carve((size_t)256 * HID * 2);
  unsigned short* Whh0r = (unsigned short*)carve((size_t)256 * HID * 2);
  unsigned short* Whh1f = (unsigned short*)carve((size_t)256 * HID * 2);
  unsigned short* Whh1r = (unsigned short*)carve((size_t)256 * HID * 2);
  float*          bias0 = (float*)carve(512 * 4);
  float*          bias1 = (float*)carve(512 * 4);
  float*          last  = (float*)carve((size_t)BATCH * 128 * 4);
  (void)ws_size; (void)n_in; (void)in_sizes; (void)out_size;

  // 1) weight packing (fp32 -> bf16, K-padding for layer 0 input weights)
  pack_bf16<<<256, K0PAD, 0, stream>>>(w_ih_l0f, W0,               300, K0PAD);
  pack_bf16<<<256, K0PAD, 0, stream>>>(w_ih_l0r, W0 + 256 * K0PAD, 300, K0PAD);
  pack_bf16<<<256, K1,    0, stream>>>(w_ih_l1f, W1,               K1,  K1);
  pack_bf16<<<256, K1,    0, stream>>>(w_ih_l1r, W1 + 256 * K1,    K1,  K1);
  pack_bf16<<<256, HID,   0, stream>>>(w_hh_l0f, Whh0f,            HID, HID);
  pack_bf16<<<256, HID,   0, stream>>>(w_hh_l0r, Whh0r,            HID, HID);
  pack_bf16<<<256, HID,   0, stream>>>(w_hh_l1f, Whh1f,            HID, HID);
  pack_bf16<<<256, HID,   0, stream>>>(w_hh_l1r, Whh1r,            HID, HID);
  concat_bias<<<2, 256, 0, stream>>>(b_l0f, b_l0r, bias0);
  concat_bias<<<2, 256, 0, stream>>>(b_l1f, b_l1r, bias1);

  // 2) embedding gather -> bf16 X0 (65536 x 320)
  embed_gather<<<MTOT, K0PAD, 0, stream>>>(x, emb, X0);

  // 3) layer-0 gate GEMM: GX = X0 * W0^T + bias0  (65536 x 512)
  wmma_gemm_bias<<<dim3(NGATES / 256, MTOT / 16), 128, 0, stream>>>(X0, W0, bias0, GXb, K0PAD, NGATES);

  // 4) layer-0 scans, both directions concurrently; store all h into X1 bf16
  lstm_scan_dual<<<dim3(BATCH / 16, 2), 128, 0, stream>>>(GXb, Whh0f, Whh0r, X1, nullptr,
                                                          SEQ_T, SEQ_T);

  // 5) layer-1 gate GEMM: GX = X1 * W1^T + bias1  (reuse gate buffer)
  wmma_gemm_bias<<<dim3(NGATES / 256, MTOT / 16), 128, 0, stream>>>(X1, W1, bias1, GXb, K1, NGATES);

  // 6) layer-1 scans concurrently: forward needs final h only; reverse's
  //    h[:, -1] is its first scan step (h0=c0=0), so one step suffices.
  lstm_scan_dual<<<dim3(BATCH / 16, 2), 128, 0, stream>>>(GXb, Whh1f, Whh1r, nullptr, last,
                                                          SEQ_T, 1);

  // 7) final FC
  fc_kernel<<<(BATCH * 12 + 127) / 128, 128, 0, stream>>>(last, fc_w, fc_b, out);
}
